// MultiHeadAttention_9749575762704
// MI455X (gfx1250) — matrix operand order repaired and checked by reading, not compiled, not run on hardware
//
#include <hip/hip_runtime.h>
#include <hip/hip_bf16.h>

// ---------------------------------------------------------------------------
// MHA forward, CDNA5 (gfx1250, wave32).
// GEMMs: v_wmma_f32_16x16x32_bf16.  LDS staging: Tensor Data Mover (TDM).
// B=2, S=2048, D=1024, H=16, hs=64.
// ---------------------------------------------------------------------------

typedef __bf16 bf16;
typedef __attribute__((ext_vector_type(4)))  __bf16 bf16x4;
typedef __attribute__((ext_vector_type(8)))  __bf16 bf16x8;
typedef __attribute__((ext_vector_type(16))) __bf16 bf16x16;
typedef __attribute__((ext_vector_type(8)))  float  f32x8;
typedef __attribute__((ext_vector_type(4)))  unsigned int u32x4;
typedef __attribute__((ext_vector_type(8)))  int i32x8;
typedef __attribute__((ext_vector_type(4)))  int i32x4;

#define S_LEN 2048
#define DM    1024
#define HEADS 16
#define HS    64
// softmax scale (1/sqrt(64)) * log2(e), folded into Q at store; softmax uses exp2
#define QSCALE (0.125f * 1.44269504088896340736f)

struct BF16x16pair { bf16x8 lo, hi; };

// A-fragment (16x32 bf16, MxK). lane<16: K 0..15; lane>=16: K 16..31
__device__ __forceinline__ bf16x16 load_a_frag(const bf16* row, int hh) {
    BF16x16pair r;
    r.lo = *(const bf16x8*)(row + hh * 16);
    r.hi = *(const bf16x8*)(row + hh * 16 + 8);
    return __builtin_bit_cast(bf16x16, r);
}

// B-fragment (32x16 bf16, KxN). lane n<16: K=0..15 of column n; lane n+16: K=16..31.
__device__ __forceinline__ bf16x16 load_b_frag16(const bf16* p) {
    BF16x16pair r;
    r.lo = *(const bf16x8*)(p);
    r.hi = *(const bf16x8*)(p + 8);
    return __builtin_bit_cast(bf16x16, r);
}

__device__ __forceinline__ f32x8 wmma_bf16(bf16x16 a, bf16x16 b, f32x8 c) {
    return __builtin_amdgcn_wmma_f32_16x16x32_bf16(
        /*neg_a=*/false, a, /*neg_b=*/false, b,
        /*c_mod=*/(short)0, c, /*reuse_a=*/false, /*reuse_b=*/false);
}

// ---------------------------------------------------------------------------
// Tensor Data Mover: 2-D tile (bf16) global -> LDS, with LDS row padding.
// Descriptor per CDNA5 ISA.  Tracked by TENSORcnt.
//   pad_interval code: 2^(code+1) DWORDs between pads; pad_amount code: code+1 DWORDs.
// ---------------------------------------------------------------------------
__device__ __forceinline__ void tdm_load_2d_bf16(
    const bf16* lds_dst, const bf16* gsrc,
    unsigned tensor_d0, unsigned tensor_d1, unsigned stride0,
    unsigned tile_d0, unsigned tile_d1,
    unsigned pad_interval_code, unsigned pad_amount_code)
{
    unsigned long long ga = (unsigned long long)gsrc;
    unsigned lds_off = (unsigned)(unsigned long long)lds_dst;  // LDS aperture low 32

    u32x4 g0;
    g0[0] = 1u;                                   // count=1, user mode, no gather
    g0[1] = lds_off;                              // lds_addr
    g0[2] = (unsigned)(ga & 0xFFFFFFFFu);         // global_addr[31:0]
    g0[3] = (unsigned)((ga >> 32) & 0x01FFFFFFu)  // global_addr[56:32]
          | (2u << 30);                           // type = 2 ("image")

    i32x8 g1;
    g1[0] = (int)((1u << 16)                      // data_size = 1 -> 2 bytes
          | (1u << 20)                            // pad_enable
          | (pad_interval_code << 22)
          | (pad_amount_code << 25));
    g1[1] = (int)((tensor_d0 & 0xFFFFu) << 16);               // tensor_dim0[15:0] @bit48
    g1[2] = (int)(((tensor_d0 >> 16) & 0xFFFFu)               // tensor_dim0[31:16]
          | ((tensor_d1 & 0xFFFFu) << 16));                   // tensor_dim1[15:0]
    g1[3] = (int)(((tensor_d1 >> 16) & 0xFFFFu)               // tensor_dim1[31:16]
          | ((tile_d0 & 0xFFFFu) << 16));                     // tile_dim0
    g1[4] = (int)(tile_d1 & 0xFFFFu);                         // tile_dim1 (tile_dim2=0)
    g1[5] = (int)stride0;                                     // tensor_dim0_stride[31:0]
    g1[6] = 0;
    g1[7] = 0;

    i32x4 gz = {0, 0, 0, 0};                       // groups 2/3 unused (2-D tensor)
#if __has_include(<hip/amd_detail/amd_gfx1250_TDM.h>)
    i32x8 gz8 = {0, 0, 0, 0, 0, 0, 0, 0};
    __builtin_amdgcn_tensor_load_to_lds(g0, g1, gz, gz, gz8, 0);
#else
    __builtin_amdgcn_tensor_load_to_lds(g0, g1, gz, gz, 0);
#endif
}

// ---------------------------------------------------------------------------
// fp32 -> bf16 (4 elements / thread)
// ---------------------------------------------------------------------------
__global__ void cvt_f32_bf16(const float* __restrict__ src, bf16* __restrict__ dst, int n4) {
    int i = blockIdx.x * blockDim.x + threadIdx.x;
    if (i < n4) {
        float4 v = ((const float4*)src)[i];
        bf16x4 o;
        o[0] = (bf16)v.x; o[1] = (bf16)v.y; o[2] = (bf16)v.z; o[3] = (bf16)v.w;
        ((bf16x4*)dst)[i] = o;
    }
}

// ---------------------------------------------------------------------------
// QKV projection.  grid = (S/128, H, 2*3).  block = 256 (8 waves).
// x chunk staged via TDM; W chunk staged transposed via vector ops.
// Q,K stored row-major [b][h][s][64]; V stored transposed [b][h][64][s].
// ---------------------------------------------------------------------------
__global__ void __launch_bounds__(256)
qkv_kernel(const bf16* __restrict__ xb,
           const bf16* __restrict__ wqb, const bf16* __restrict__ wkb,
           const bf16* __restrict__ wvb,
           const float* __restrict__ bq, const float* __restrict__ bk,
           const float* __restrict__ bv,
           bf16* __restrict__ qb, bf16* __restrict__ kb, bf16* __restrict__ vtb) {
    const int mtile = blockIdx.x;
    const int h     = blockIdx.y;
    const int b     = blockIdx.z / 3;
    const int mat   = blockIdx.z % 3;

    const int tid  = threadIdx.x;
    const int lane = tid & 31;
    const int wave = tid >> 5;
    const int hh   = lane >> 4;
    const int l15  = lane & 15;

    const bf16*  W    = (mat == 0) ? wqb : (mat == 1) ? wkb : wvb;
    const float* bias = (mat == 0) ? bq  : (mat == 1) ? bk  : bv;
    W += (size_t)h * DM * HS;

    __shared__ bf16 xs[128][40];   // x chunk  (128 rows x 32 K), rows padded 32->40
    __shared__ bf16 wsT[64][40];   // W chunk transposed (64 f x 32 K), padded

    const bf16* xrow = xb + ((size_t)b * S_LEN + (size_t)mtile * 128) * DM;

    f32x8 acc[4] = {};

    for (int k0 = 0; k0 < DM; k0 += 32) {
        __syncthreads();                       // everyone done with previous chunk
        if (wave == 0) {
            // 128x32 tile of x -> xs; row = 32 elem = 16 dwords (code 3),
            // pad 8 elem = 4 dwords (code 3)
            tdm_load_2d_bf16(&xs[0][0], xrow + k0,
                             /*tensor_d0=*/DM, /*tensor_d1=*/128, /*stride0=*/DM,
                             /*tile_d0=*/32, /*tile_d1=*/128,
                             /*pad_interval=*/3, /*pad_amount=*/3);
        }
        {   // stage W chunk transposed: [d][f] -> wsT[f][d]
            int dd = tid >> 3, f0 = (tid & 7) * 8;
            bf16x8 v = *(const bf16x8*)(W + (size_t)(k0 + dd) * HS + f0);
#pragma unroll
            for (int i = 0; i < 8; ++i) wsT[f0 + i][dd] = v[i];
        }
        if (wave == 0) __builtin_amdgcn_s_wait_tensorcnt(0);
        __syncthreads();

        bf16x16 afrag = load_a_frag(&xs[wave * 16 + l15][0], hh);
#pragma unroll
        for (int n = 0; n < 4; ++n) {
            bf16x16 bfrag = load_b_frag16(&wsT[n * 16 + l15][hh * 16]);
            acc[n] = wmma_bf16(afrag, bfrag, acc[n]);
        }
    }

    // epilogue: lane holds column f, rows s = sbase + hh*8 + i
    const int sbase = mtile * 128 + wave * 16;
    const size_t head = ((size_t)b * HEADS + h);
#pragma unroll
    for (int n = 0; n < 4; ++n) {
        int f = n * 16 + l15;
        float bvv = bias[h * HS + f];
        f32x8 c = acc[n];
#pragma unroll
        for (int i = 0; i < 8; ++i) c[i] += bvv;

        if (mat == 0) {                 // Q row-major, pre-scaled
#pragma unroll
            for (int i = 0; i < 8; ++i) {
                int s = sbase + hh * 8 + i;
                qb[(head * S_LEN + s) * HS + f] = (bf16)(c[i] * QSCALE);
            }
        } else if (mat == 1) {          // K row-major
#pragma unroll
            for (int i = 0; i < 8; ++i) {
                int s = sbase + hh * 8 + i;
                kb[(head * S_LEN + s) * HS + f] = (bf16)c[i];
            }
        } else {                        // V transposed [64][S] -> contiguous store
            bf16x8 v;
#pragma unroll
            for (int i = 0; i < 8; ++i) v[i] = (bf16)c[i];
            *(bf16x8*)&vtb[(head * HS + f) * S_LEN + sbase + hh * 8] = v;
        }
    }
}

// ---------------------------------------------------------------------------
// Flash attention (transposed tiles).  grid = (S/128, H, B).  block = 256.
// K / V^T chunks double-buffered in LDS, filled by TDM (wave 0 issues DMA,
// TENSORcnt synchronizes, one workgroup barrier per chunk).
// S^T[t][q] = sum_e K[t][e]*Q[q][e]   (Q pre-scaled, log2 domain)
// ctx^T[f][q] += sum_t V[t][f]*P[t][q]
// ---------------------------------------------------------------------------
__global__ void __launch_bounds__(256)
attn_kernel(const bf16* __restrict__ qb, const bf16* __restrict__ kb,
            const bf16* __restrict__ vtb, bf16* __restrict__ ctxb) {
    const int mtile = blockIdx.x;
    const int h     = blockIdx.y;
    const int b     = blockIdx.z;

    const int tid  = threadIdx.x;
    const int lane = tid & 31;
    const int wave = tid >> 5;
    const int hh   = lane >> 4;
    const int l15  = lane & 15;

    __shared__ bf16 ks[2][32][72];     // K chunk  (32 keys x 64 e), rows 64->72
    __shared__ bf16 vts[2][64][40];    // V^T chunk (64 f x 32 keys), rows 32->40
    __shared__ bf16 ps[8][16][40];     // per-wave P (16 q x 32 keys), padded

    const size_t head = ((size_t)b * HEADS + h) * (size_t)S_LEN * HS;
    const bf16* Qh  = qb  + head;
    const bf16* Kh  = kb  + head;
    const bf16* Vth = vtb + head;      // [64][2048]

    const int q = mtile * 128 + wave * 16 + l15;   // this lane's query

    // preload Q B-fragments (two e-halves of 32), contiguous 32B per lane
    bf16x16 qf[2];
#pragma unroll
    for (int eh = 0; eh < 2; ++eh)
        qf[eh] = load_b_frag16(Qh + (size_t)q * HS + eh * 32 + hh * 16);

    f32x8 ctx[4] = {};
    float mrun = -1e30f, lrun = 0.0f;

    // prologue DMA: chunk 0 into buffer 0
    if (wave == 0) {
        // K tile: row = 64 elem = 32 dwords (code 4), pad 4 dwords (code 3)
        tdm_load_2d_bf16(&ks[0][0][0], Kh,
                         HS, S_LEN, HS, /*tile*/ HS, 32, 4, 3);
        // V^T tile: row = 32 elem = 16 dwords (code 3), pad 4 dwords (code 3)
        tdm_load_2d_bf16(&vts[0][0][0], Vth,
                         S_LEN, HS, S_LEN, /*tile*/ 32, HS, 3, 3);
    }

    const int NIT = S_LEN / 32;
    for (int it = 0; it < NIT; ++it) {
        const int buf = it & 1;
        if (wave == 0) __builtin_amdgcn_s_wait_tensorcnt(0);
        __syncthreads();               // chunk `it` resident; prev compute done
        if (wave == 0 && it + 1 < NIT) {
            const int t1 = (it + 1) * 32;
            tdm_load_2d_bf16(&ks[buf ^ 1][0][0], Kh + (size_t)t1 * HS,
                             HS, S_LEN, HS, HS, 32, 4, 3);
            tdm_load_2d_bf16(&vts[buf ^ 1][0][0], Vth + t1,
                             S_LEN, HS, S_LEN, 32, HS, 3, 3);
        }

        // ---- scores S^T: 32 keys x 16 queries (two 16x16 tiles) ----
        f32x8 c0t = {}, c1t = {};
#pragma unroll
        for (int eh = 0; eh < 2; ++eh) {
            bf16x16 a0 = load_a_frag(&ks[buf][l15][eh * 32], hh);
            bf16x16 a1 = load_a_frag(&ks[buf][16 + l15][eh * 32], hh);
            c0t = wmma_bf16(a0, qf[eh], c0t);
            c1t = wmma_bf16(a1, qf[eh], c1t);
        }

        // ---- online softmax (log2 domain), per-query stats live in lanes ----
        float lm = c0t[0];
#pragma unroll
        for (int i = 0; i < 8; ++i) { lm = fmaxf(lm, c0t[i]); lm = fmaxf(lm, c1t[i]); }
        lm = fmaxf(lm, __shfl_xor(lm, 16, 32));
        float mnew  = fmaxf(mrun, lm);
        float alpha = __builtin_exp2f(mrun - mnew);
        float rsum  = 0.0f;
#pragma unroll
        for (int i = 0; i < 8; ++i) {
            c0t[i] = __builtin_exp2f(c0t[i] - mnew); rsum += c0t[i];
            c1t[i] = __builtin_exp2f(c1t[i] - mnew); rsum += c1t[i];
        }
        rsum += __shfl_xor(rsum, 16, 32);
        lrun = lrun * alpha + rsum;
        mrun = mnew;
#pragma unroll
        for (int n = 0; n < 4; ++n)
#pragma unroll
            for (int i = 0; i < 8; ++i) ctx[n][i] *= alpha;

        // ---- P -> bf16 via wave-private LDS to get B-fragment layout ----
        bf16x8 p0, p1;
#pragma unroll
        for (int i = 0; i < 8; ++i) { p0[i] = (bf16)c0t[i]; p1[i] = (bf16)c1t[i]; }
        *(bf16x8*)&ps[wave][l15][hh * 8]      = p0;   // keys t0 + {0..7 | 8..15}
        *(bf16x8*)&ps[wave][l15][16 + hh * 8] = p1;   // keys t0 + {16..23 | 24..31}
        bf16x16 pf = load_b_frag16(&ps[wave][l15][hh * 16]);

        // ---- ctx^T accumulate: 4 feature tiles, K=32 keys ----
#pragma unroll
        for (int n = 0; n < 4; ++n) {
            bf16x16 af = load_a_frag(&vts[buf][n * 16 + l15][0], hh);
            ctx[n] = wmma_bf16(af, pf, ctx[n]);
        }
    }

    // epilogue: normalize, store ctx as [b][s][h*64 + f] bf16 (contiguous 16B/lane)
    float inv = 1.0f / lrun;
#pragma unroll
    for (int n = 0; n < 4; ++n) {
        bf16x8 v;
#pragma unroll
        for (int i = 0; i < 8; ++i) v[i] = (bf16)(ctx[n][i] * inv);
        size_t off = ((size_t)b * S_LEN + q) * DM + h * HS + n * 16 + hh * 8;
        *(bf16x8*)&ctxb[off] = v;
    }
}

// ---------------------------------------------------------------------------
// Output projection (transposed orientation).  grid = (4096/128, 1024/64).
// out[s][f] = sum_d ctx[s][d] * Wo[d][f] + bo[f]   (fp32 out)
// ---------------------------------------------------------------------------
__global__ void __launch_bounds__(256)
proj_kernel(const bf16* __restrict__ ctxb, const bf16* __restrict__ wob,
            const float* __restrict__ bo, float* __restrict__ out) {
    const int stile = blockIdx.x;          // 128 rows of (B*S)=4096
    const int ftile = blockIdx.y;          // 64 output features

    const int tid  = threadIdx.x;
    const int lane = tid & 31;
    const int wave = tid >> 5;
    const int hh   = lane >> 4;
    const int l15  = lane & 15;

    __shared__ bf16 wsT[64][40];           // Wo chunk transposed (64 f x 32 K)

    const int srow = stile * 128 + wave * 16 + l15;   // this lane's output row

    f32x8 acc[4] = {};
    for (int k0 = 0; k0 < DM; k0 += 32) {
        __syncthreads();
        {   // stage Wo chunk transposed
            int dd = tid >> 3, f0 = (tid & 7) * 8;
            bf16x8 v = *(const bf16x8*)(wob + (size_t)(k0 + dd) * DM + ftile * 64 + f0);
#pragma unroll
            for (int i = 0; i < 8; ++i) wsT[f0 + i][dd] = v[i];
        }
        __syncthreads();

        // B-fragment straight from global ctx: contiguous 32B per lane
        bf16x16 bfrag = load_b_frag16(ctxb + (size_t)srow * DM + k0 + hh * 16);
#pragma unroll
        for (int n = 0; n < 4; ++n) {
            bf16x16 af = load_a_frag(&wsT[n * 16 + l15][0], hh);
            acc[n] = wmma_bf16(af, bfrag, acc[n]);
        }
    }

    // epilogue: lane holds 8 consecutive features per tile for row srow
#pragma unroll
    for (int n = 0; n < 4; ++n) {
        int f = ftile * 64 + n * 16 + hh * 8;
        f32x8 bias = *(const f32x8*)(bo + f);
        f32x8 r = acc[n];
#pragma unroll
        for (int i = 0; i < 8; ++i) r[i] += bias[i];
        *(f32x8*)(out + (size_t)srow * DM + f) = r;
    }
}

// ---------------------------------------------------------------------------
// launch
// ---------------------------------------------------------------------------
extern "C" void kernel_launch(void* const* d_in, const int* in_sizes, int n_in,
                              void* d_out, int out_size, void* d_ws, size_t ws_size,
                              hipStream_t stream) {
    (void)in_sizes; (void)n_in; (void)out_size; (void)ws_size;

    const float* x  = (const float*)d_in[0];
    const float* Wq = (const float*)d_in[1];
    const float* bq = (const float*)d_in[2];
    const float* Wk = (const float*)d_in[3];
    const float* bk = (const float*)d_in[4];
    const float* Wv = (const float*)d_in[5];
    const float* bv = (const float*)d_in[6];
    const float* Wo = (const float*)d_in[7];
    const float* bo = (const float*)d_in[8];
    float* out = (float*)d_out;

    char* ws = (char*)d_ws;
    const size_t MB = (size_t)1 << 20;
    bf16* xb   = (bf16*)(ws);              //  8 MB: x bf16       [2][2048][1024]
    bf16* wqb  = (bf16*)(ws + 8  * MB);    //  2 MB
    bf16* wkb  = (bf16*)(ws + 10 * MB);    //  2 MB
    bf16* wvb  = (bf16*)(ws + 12 * MB);    //  2 MB
    bf16* wob  = (bf16*)(ws + 14 * MB);    //  2 MB
    bf16* qb   = (bf16*)(ws + 16 * MB);    //  8 MB: Q (scaled)   [2][16][2048][64]
    bf16* kb   = (bf16*)(ws + 24 * MB);    //  8 MB: K            [2][16][2048][64]
    bf16* vtb  = (bf16*)(ws + 32 * MB);    //  8 MB: V^T          [2][16][64][2048]
    bf16* ctxb = (bf16*)(ws + 40 * MB);    //  8 MB: ctx          [2][2048][1024]

    const int NX = 2 * S_LEN * DM;               // 4,194,304
    const int NW = HEADS * DM * HS;              // 1,048,576 (also Wo)
    cvt_f32_bf16<<<(NX / 4 + 255) / 256, 256, 0, stream>>>(x,  xb,  NX / 4);
    cvt_f32_bf16<<<(NW / 4 + 255) / 256, 256, 0, stream>>>(Wq, wqb, NW / 4);
    cvt_f32_bf16<<<(NW / 4 + 255) / 256, 256, 0, stream>>>(Wk, wkb, NW / 4);
    cvt_f32_bf16<<<(NW / 4 + 255) / 256, 256, 0, stream>>>(Wv, wvb, NW / 4);
    cvt_f32_bf16<<<(NW / 4 + 255) / 256, 256, 0, stream>>>(Wo, wob, NW / 4);

    qkv_kernel <<<dim3(S_LEN / 128, HEADS, 2 * 3), 256, 0, stream>>>(
        xb, wqb, wkb, wvb, bq, bk, bv, qb, kb, vtb);
    attn_kernel<<<dim3(S_LEN / 128, HEADS, 2),     256, 0, stream>>>(
        qb, kb, vtb, ctxb);
    proj_kernel<<<dim3(2 * S_LEN / 128, DM / 64),  256, 0, stream>>>(
        ctxb, wob, bo, out);
}
